// TemporalCrossAttention_44985487458306
// MI455X (gfx1250) — matrix operand order repaired and checked by reading, not compiled, not run on hardware
//
#include <hip/hip_runtime.h>
#include <hip/hip_bf16.h>

// ---------------------------------------------------------------------------
// TemporalCrossAttention for MI455X (gfx1250), wave32 + v_wmma_f32_16x16x32_bf16
//   B=2, S=2048, E=512, H=8, Dh=64
// Pipeline:
//   proj<f32,0>  : q = query@Wq^T+bq  -> bf16 [BH,S,Dh]
//   proj<f32,0>  : k = key  @Wk^T+bk  -> bf16 [BH,S,Dh]
//   proj<f32,1>  : vT= (value@Wv^T+bv)^T -> bf16 [BH,Dh,S]
//   attn         : scores->bias->softmax->attn_weights (d_out) + ctx bf16 (ws)
//   proj<bf16,2> : attn_output = ctx@Wo^T+bo -> f32 (d_out)
// ---------------------------------------------------------------------------

typedef __bf16 bf16_t;
typedef __attribute__((ext_vector_type(16))) __bf16 v16bf;
typedef __attribute__((ext_vector_type(8)))  float  v8f;

#define B_   2
#define S_   2048
#define E_   512
#define H_   8
#define DH_  64
#define SCALE_ 0.125f   // 64^-0.5

// ---- WMMA fragment helpers (contiguous k order, wave32) -------------------------
// A (16x32, 16-bit): lanes 0-15 -> M=lane, K run [0..15];
//                    lanes 16-31 -> M=lane-16, K run [16..31].
template <typename T>
__device__ inline v16bf load_a_frag(const T* __restrict__ base, int ld,
                                    int m0, int k0, int lane) {
  const int m  = m0 + (lane & 15);
  const int kb = k0 + ((lane >> 4) << 4);
  const T* p0 = base + (size_t)m * ld + kb;
  const T* p1 = p0 + 8;
  v16bf a;
#pragma unroll
  for (int i = 0; i < 8; ++i) a[i]     = (__bf16)(float)p0[i];
#pragma unroll
  for (int i = 0; i < 8; ++i) a[8 + i] = (__bf16)(float)p1[i];
  return a;
}

// B (32x16, 16-bit): lane -> N=lane&15; K run of 16 starting at 16*(lane>=16).
// Element (K,N) comes from src[(n0+N)*ld + k0 + K]  (i.e. src is N-major).
template <typename T>
__device__ inline v16bf load_b_frag(const T* __restrict__ base, int ld,
                                    int n0, int k0, int lane) {
  const int n  = n0 + (lane & 15);
  const int kb = k0 + ((lane >> 4) << 4);
  const T* p = base + (size_t)n * ld + kb;
  v16bf b;
#pragma unroll
  for (int i = 0; i < 16; ++i) b[i] = (__bf16)(float)p[i];
  return b;
}

__device__ inline v8f wmma_bf16(v16bf a, v16bf b, v8f c) {
  return __builtin_amdgcn_wmma_f32_16x16x32_bf16(
      /*neg_a=*/false, a, /*neg_b=*/false, b,
      /*c_mod=*/(short)0, c, /*reuse_a=*/false, /*reuse_b=*/false);
}

// ---- Projection GEMM: Y[4096x512] = X[4096x512] @ W^T + bias ---------------
// One wave per 16x16 output tile; 8 waves per block; K looped in steps of 32.
// MODE 0: out bf16 head-major [BH,S,Dh]     (q, k)
// MODE 1: out bf16 transposed  [BH,Dh,S]    (vT)
// MODE 2: out f32 row-major    [B*S, E]     (final attn_output)
template <typename TIn, int MODE>
__global__ __launch_bounds__(256) void proj_kernel(
    const TIn* __restrict__ X, const float* __restrict__ W,
    const float* __restrict__ bias, void* __restrict__ Yv) {
  const int lane = threadIdx.x & 31;
  const int wid  = threadIdx.x >> 5;
  const int gw   = blockIdx.x * 8 + wid;
  const int ntn  = E_ / 16;                 // 32 N-tiles
  const int m0   = (gw / ntn) * 16;
  const int n0   = (gw % ntn) * 16;

  v8f acc = {};
#pragma unroll 4
  for (int k0 = 0; k0 < E_; k0 += 32) {
    v16bf a = load_a_frag<TIn>(X, E_, m0, k0, lane);     // A = X tile
    v16bf b = load_b_frag<float>(W, E_, n0, k0, lane);   // B[k][n] = W[n][k]
    acc = wmma_bf16(a, b, acc);
  }

  const int n  = n0 + (lane & 15);
  const float bv = bias[n];
#pragma unroll
  for (int r = 0; r < 8; ++r) {
    const int m = m0 + r + ((lane >> 4) << 3);
    const float y = acc[r] + bv;
    if (MODE == 2) {
      ((float*)Yv)[(size_t)m * E_ + n] = y;
    } else {
      const int bb = m >> 11, s = m & (S_ - 1);
      const int h = n >> 6, dh = n & 63;
      if (MODE == 0)
        ((bf16_t*)Yv)[(((size_t)(bb * H_ + h)) * S_ + s) * DH_ + dh] = (__bf16)y;
      else
        ((bf16_t*)Yv)[(((size_t)(bb * H_ + h)) * DH_ + dh) * S_ + s] = (__bf16)y;
    }
  }
}

// ---- Fused attention: one block per (bh, 16-query tile) --------------------
__global__ __launch_bounds__(256) void attn_kernel(
    const bf16_t* __restrict__ q, const bf16_t* __restrict__ k,
    const bf16_t* __restrict__ vT, const float* __restrict__ dist,
    const int* __restrict__ mask, const float* __restrict__ tbias,
    float* __restrict__ attn_w, bf16_t* __restrict__ ctx) {
  extern __shared__ char smem[];
  bf16_t* probs  = (bf16_t*)smem;                              // 16 x 2048 bf16 (64 KB)
  float* partial = (float*)(smem + (size_t)16 * S_ * 2);       // 8 x 16 x 64 f32 (32 KB)
  float* rowinv  = partial + 8 * 16 * DH_;                     // 16 f32

  const int tid  = threadIdx.x;
  const int lane = tid & 31;
  const int wid  = tid >> 5;
  const int bh   = blockIdx.x >> 7;     // 128 q-tiles per (b,h)
  const int q0   = (blockIdx.x & 127) * 16;
  const int h    = bh & (H_ - 1);
  const float tb = tbias[h];

  const bf16_t* qb = q  + (size_t)bh * S_ * DH_;
  const bf16_t* kb = k  + (size_t)bh * S_ * DH_;
  const bf16_t* vb = vT + (size_t)bh * DH_ * S_;

  const int Mofs = (lane >> 4) << 3;    // accumulator row base (0 or 8)
  const int N    = lane & 15;

  // ---- Phase 1: scores = (Q K^T)*scale + tb*exp(-dist), masked -> LDS ----
  v16bf a0 = load_a_frag<bf16_t>(qb, DH_, q0, 0,  lane);
  v16bf a1 = load_a_frag<bf16_t>(qb, DH_, q0, 32, lane);
  for (int kt = 0; kt < 16; ++kt) {     // this wave covers keys [wid*256, +256)
    const int n0 = wid * 256 + kt * 16;
    v16bf b0 = load_b_frag<bf16_t>(kb, DH_, n0, 0,  lane);
    v16bf b1 = load_b_frag<bf16_t>(kb, DH_, n0, 32, lane);
    v8f acc = {};
    acc = wmma_bf16(a0, b0, acc);
    acc = wmma_bf16(a1, b1, acc);
#pragma unroll
    for (int r = 0; r < 8; ++r) {
      const int row = Mofs + r;
      const int col = n0 + N;
      const size_t gij = (size_t)(q0 + row) * S_ + col;
      float sv = acc[r] * SCALE_ + tb * __expf(-dist[gij]);
      if (mask[gij] == 0) sv = -1e9f;
      probs[row * S_ + col] = (__bf16)sv;
    }
  }
  __syncthreads();

  // ---- Phase 2: row softmax (16 threads per row, shuffle reductions) -----
  {
    const int r = tid >> 4, c = tid & 15;
    bf16_t* pr = probs + r * S_;
    float mx = -3.4e38f;
    for (int j = c; j < S_; j += 16) mx = fmaxf(mx, (float)pr[j]);
#pragma unroll
    for (int o = 1; o < 16; o <<= 1) mx = fmaxf(mx, __shfl_xor(mx, o, 16));
    float sum = 0.f;
    for (int j = c; j < S_; j += 16) {
      const float e = __expf((float)pr[j] - mx);
      pr[j] = (__bf16)e;
      sum += e;
    }
#pragma unroll
    for (int o = 1; o < 16; o <<= 1) sum += __shfl_xor(sum, o, 16);
    if (c == 0) rowinv[r] = 1.f / sum;
  }
  __syncthreads();

  // ---- Phase 2b: normalize + coalesced attn_weights store ----------------
  {
    float* wout = attn_w + ((size_t)bh * S_ + q0) * S_;   // row stride == S_
    for (int idx = tid; idx < 16 * S_; idx += 256) {
      const float p = (float)probs[idx] * rowinv[idx >> 11];
      probs[idx] = (__bf16)p;
      wout[idx] = p;
    }
  }
  __syncthreads();

  // ---- Phase 3: ctx tile = P @ V (per-wave partials over 256 keys) -------
  v8f acc[4] = {};
  for (int ks = 0; ks < 8; ++ks) {
    const int k0 = wid * 256 + ks * 32;
    v16bf a = load_a_frag<bf16_t>(probs, S_, 0, k0, lane);   // from LDS
#pragma unroll
    for (int d = 0; d < 4; ++d) {
      v16bf b = load_b_frag<bf16_t>(vb, S_, d * 16, k0, lane); // V^T: contiguous
      acc[d] = wmma_bf16(a, b, acc[d]);
    }
  }
#pragma unroll
  for (int d = 0; d < 4; ++d)
#pragma unroll
    for (int r = 0; r < 8; ++r)
      partial[(wid * 16 + Mofs + r) * DH_ + d * 16 + N] = acc[d][r];
  __syncthreads();

  // ---- Cross-wave reduction -> ctx (bf16, [B,S,E] row-major) -------------
  {
    const int bb = bh >> 3;
    for (int idx = tid; idx < 16 * DH_; idx += 256) {
      const int row = idx >> 6, col = idx & 63;
      float s = 0.f;
#pragma unroll
      for (int w = 0; w < 8; ++w) s += partial[w * 16 * DH_ + idx];
      ctx[((size_t)(bb * S_ + q0 + row)) * E_ + h * DH_ + col] = (__bf16)s;
    }
  }
}

// ---------------------------------------------------------------------------
extern "C" void kernel_launch(void* const* d_in, const int* in_sizes, int n_in,
                              void* d_out, int out_size, void* d_ws, size_t ws_size,
                              hipStream_t stream) {
  (void)in_sizes; (void)n_in; (void)out_size; (void)ws_size;

  const float* query = (const float*)d_in[0];
  const float* key_  = (const float*)d_in[1];
  const float* value = (const float*)d_in[2];
  const int*   tmask = (const int*)  d_in[3];
  const float* ldist = (const float*)d_in[4];
  const float* Wq = (const float*)d_in[5];
  const float* bq = (const float*)d_in[6];
  const float* Wk = (const float*)d_in[7];
  const float* bk = (const float*)d_in[8];
  const float* Wv = (const float*)d_in[9];
  const float* bv = (const float*)d_in[10];
  const float* Wo = (const float*)d_in[11];
  const float* bo = (const float*)d_in[12];
  const float* tbias = (const float*)d_in[13];

  float* attn_out = (float*)d_out;                              // B*S*E
  float* attn_w   = attn_out + (size_t)B_ * S_ * E_;            // B*H*S*S

  const size_t headElems = (size_t)B_ * H_ * S_ * DH_;          // 2,097,152
  bf16_t* qd  = (bf16_t*)d_ws;
  bf16_t* kd  = qd  + headElems;
  bf16_t* vTd = kd  + headElems;
  bf16_t* ctx = vTd + headElems;                                // 16 MB total ws

  const dim3 blk(256);
  const int projBlocks = (B_ * S_ / 16) * (E_ / 16) / 8;        // 1024

  proj_kernel<float, 0><<<projBlocks, blk, 0, stream>>>(query, Wq, bq, qd);
  proj_kernel<float, 0><<<projBlocks, blk, 0, stream>>>(key_,  Wk, bk, kd);
  proj_kernel<float, 1><<<projBlocks, blk, 0, stream>>>(value, Wv, bv, vTd);

  const size_t smem = (size_t)16 * S_ * sizeof(bf16_t)          // probs
                    + (size_t)8 * 16 * DH_ * sizeof(float)      // partials
                    + 16 * sizeof(float);                       // rowinv
  hipFuncSetAttribute((const void*)attn_kernel,
                      hipFuncAttributeMaxDynamicSharedMemorySize, (int)smem);
  attn_kernel<<<B_ * H_ * (S_ / 16), blk, smem, stream>>>(
      qd, kd, vTd, ldist, tmask, tbias, attn_w, ctx);

  proj_kernel<bf16_t, 2><<<projBlocks, blk, 0, stream>>>(ctx, Wo, bo, attn_out);
}
